// Linear_fft_2d_1932735284063
// MI455X (gfx1250) — matrix operand order repaired and checked by reading, not compiled, not run on hardware
//
#include <hip/hip_runtime.h>
#include <hip/hip_bf16.h>
#include <math.h>

// Problem constants (match the reference).
#define BATCH 2048
#define OUTD  4096
#define IND   4096
#define KH    2049            // IND/2 + 1 columns kept by irfftn
#define THRESH 8388608LL      // KEEPRATE * OUT * IN = 0.5 * 4096*4096

// GEMM tiling
#define KSTAGE 64             // K depth staged into LDS per double-buffer step
#define NSTAGE (IND / KSTAGE) // 64 stages
#define LDSW   72             // padded row stride in ushorts (144 B, 16B multiple)

typedef __attribute__((ext_vector_type(16))) __bf16         v16bf;
typedef __attribute__((ext_vector_type(8)))  float          v8f;
typedef __attribute__((ext_vector_type(8)))  unsigned short v8us;

union Frag { v8us h[2]; v16bf v; };

// float -> bf16 round-to-nearest-even
static __device__ __forceinline__ unsigned short f2bf(float f) {
  unsigned int u = __float_as_uint(f);
  u += 0x7FFFu + ((u >> 16) & 1u);
  return (unsigned short)(u >> 16);
}

// ---------------------------------------------------------------------------
// In-LDS 4096-point complex inverse FFT (sign = +1), Stockham radix-2.
// 256 threads, 8 butterflies/thread/stage; all reads land in registers before
// the barrier, so one 32 KB buffer suffices. No 1/N scaling here.
// ---------------------------------------------------------------------------
static __device__ __forceinline__ void block_ifft4096(float2* buf, int tid) {
  for (int Ns = 1; Ns < 4096; Ns <<= 1) {
    float2 ra[8], rb[8];
    int ti[8];
#pragma unroll
    for (int i = 0; i < 8; ++i) {
      int t = tid + (i << 8);                 // 0..2047
      float2 a = buf[t];
      float2 b = buf[t + 2048];
      int j = t & (Ns - 1);
      float ang = 3.14159265358979323846f * (float)j / (float)Ns; // +2*pi*j/(2*Ns)
      float s, c;
      __sincosf(ang, &s, &c);
      float2 wb = make_float2(c * b.x - s * b.y, c * b.y + s * b.x);
      ra[i] = make_float2(a.x + wb.x, a.y + wb.y);
      rb[i] = make_float2(a.x - wb.x, a.y - wb.y);
      ti[i] = ((t / Ns) * (Ns << 1)) + j;
    }
    __syncthreads();
#pragma unroll
    for (int i = 0; i < 8; ++i) {
      buf[ti[i]]      = ra[i];
      buf[ti[i] + Ns] = rb[i];
    }
    __syncthreads();
  }
}

// ---------------------------------------------------------------------------
// Kernel 1: cast activations f32 -> bf16 (vectorized 4-wide)
// ---------------------------------------------------------------------------
__global__ void cast_bf16_kernel(const float* __restrict__ in,
                                 unsigned short* __restrict__ out, int n4) {
  int i = blockIdx.x * blockDim.x + threadIdx.x;
  int stride = gridDim.x * blockDim.x;
  for (; i < n4; i += stride) {
    float4 f = ((const float4*)in)[i];
    ushort4 o;
    o.x = f2bf(f.x); o.y = f2bf(f.y); o.z = f2bf(f.z); o.w = f2bf(f.w);
    ((ushort4*)out)[i] = o;
  }
}

// ---------------------------------------------------------------------------
// Kernel 2: per-column masked inverse FFT along axis 0.
// block = column k2 in [0, 2049); output T[n1, k2] with 1/OUT scaling.
// ---------------------------------------------------------------------------
__global__ void fft_col_kernel(const float* __restrict__ wr,
                               const float* __restrict__ wi,
                               const long long* __restrict__ zmat,
                               float2* __restrict__ T) {
  __shared__ float2 buf[4096];
  const int k2 = blockIdx.x;
  const int tid = threadIdx.x;
  for (int k1 = tid; k1 < OUTD; k1 += 256) {
    size_t idx = (size_t)k1 * IND + k2;
    float m = (zmat[idx] <= THRESH) ? 1.0f : 0.0f;   // rank-keep mask (IDROP)
    buf[k1] = make_float2(wr[idx] * m, wi[idx] * m);
  }
  __syncthreads();
  block_ifft4096(buf, tid);
  const float s = 1.0f / 4096.0f;
  for (int n1 = tid; n1 < OUTD; n1 += 256) {
    float2 v = buf[n1];
    T[(size_t)n1 * KH + k2] = make_float2(v.x * s, v.y * s);
  }
}

// ---------------------------------------------------------------------------
// Kernel 3: per-row irfft along axis 1 via Hermitian extension + complex IFFT.
// block = row n1; outputs bf16 spatial weight row (K-contiguous for GEMM B).
// ---------------------------------------------------------------------------
__global__ void fft_row_kernel(const float2* __restrict__ T,
                               unsigned short* __restrict__ Wbf) {
  __shared__ float2 buf[4096];
  const int n1 = blockIdx.x;
  const int tid = threadIdx.x;
  for (int k2 = tid; k2 < KH; k2 += 256) {
    float2 x = T[(size_t)n1 * KH + k2];
    buf[k2] = x;
    if (k2 >= 1 && k2 <= 2047) buf[4096 - k2] = make_float2(x.x, -x.y); // conj sym
  }
  __syncthreads();
  block_ifft4096(buf, tid);
  const float s = 1.0f / 4096.0f;       // irfft's own 1/IN
  for (int n2 = tid; n2 < IND; n2 += 256)
    Wbf[(size_t)n1 * IND + n2] = f2bf(buf[n2].x * s);
}

// ---------------------------------------------------------------------------
// Async global->LDS copy of one 128 x KSTAGE bf16 stage for A and B.
// Per thread: 4 chunks of 16 B per matrix (8 async ops/wave/stage).
// gA/gB: per-lane global base (row r0, halfword h); lA/lB: per-lane LDS byte
// address in the destination buffer.
// ---------------------------------------------------------------------------
static __device__ __forceinline__ void async_copy_stage(
    const unsigned short* gA, const unsigned short* gB,
    unsigned int lA, unsigned int lB, int kbase) {
#pragma unroll
  for (int q = 0; q < 4; ++q) {
    const unsigned short* ga = gA + (size_t)q * 32 * IND + kbase;
    unsigned int la = lA + (unsigned int)(q * 32 * LDSW * 2);
    asm volatile("global_load_async_to_lds_b128 %0, %1, off"
                 :: "v"(la), "v"(ga) : "memory");
  }
#pragma unroll
  for (int q = 0; q < 4; ++q) {
    const unsigned short* gb = gB + (size_t)q * 32 * IND + kbase;
    unsigned int lb = lB + (unsigned int)(q * 32 * LDSW * 2);
    asm volatile("global_load_async_to_lds_b128 %0, %1, off"
                 :: "v"(lb), "v"(gb) : "memory");
  }
}

// ---------------------------------------------------------------------------
// Kernel 4: out = A(bf16) * W(bf16)^T + bias, f32 accumulate via WMMA.
// 256 threads = 8 waves; 128x128 tile; K staged 64-deep, double-buffered in
// LDS via GLOBAL_LOAD_ASYNC_TO_LDS_B128 (ASYNCcnt), per-wave 2x4 WMMA tiles.
// ---------------------------------------------------------------------------
__global__ void __launch_bounds__(256)
gemm_wmma_bf16_kernel(const unsigned short* __restrict__ Abf,
                      const unsigned short* __restrict__ Wbf,
                      const float* __restrict__ bias,
                      float* __restrict__ out) {
  __shared__ unsigned short As[2][128 * LDSW];   // 2 x 18 KB
  __shared__ unsigned short Bs[2][128 * LDSW];   // 2 x 18 KB

  const int tid  = threadIdx.x;
  const int wave = tid >> 5;
  const int lane = tid & 31;
  const int wm = (wave & 3) * 32;                // wave's M offset in tile
  const int wn = (wave >> 2) * 64;               // wave's N offset in tile
  const int m0 = blockIdx.y * 128;
  const int n0 = blockIdx.x * 128;

  // async staging map: thread -> (row r0 + q*32, 16B chunk h) of 128 x 64 tile
  const int r0 = tid >> 3;                       // 0..31
  const int h  = (tid & 7) * 8;                  // ushort offset 0..56
  const unsigned short* gA = Abf + (size_t)(m0 + r0) * IND + h;
  const unsigned short* gB = Wbf + (size_t)(n0 + r0) * IND + h;
  // low 32 bits of a generic LDS pointer == LDS byte offset (ISA 10.2)
  const unsigned int lA0 = (unsigned int)(size_t)&As[0][r0 * LDSW + h];
  const unsigned int lA1 = (unsigned int)(size_t)&As[1][r0 * LDSW + h];
  const unsigned int lB0 = (unsigned int)(size_t)&Bs[0][r0 * LDSW + h];
  const unsigned int lB1 = (unsigned int)(size_t)&Bs[1][r0 * LDSW + h];

  // fragment lane mapping (ISA 16-bit A 16x32 / B 32x16 layouts)
  const int fm = lane & 15;
  const int ka = (lane >> 4) * 8;                // A: K chunks {ka..+7, ka+16..+23}
  const int kb = (lane >> 4) * 8;                // B: K chunks {kb..+7, kb+16..+23}

  v8f acc[2][4] = {};

  // prologue: stage 0 -> buffer 0
  async_copy_stage(gA, gB, lA0, lB0, 0);

  for (int s = 0; s < NSTAGE; ++s) {
    const int cur = s & 1;
    if (s + 1 < NSTAGE) {
      // stream next stage into the spare buffer while we compute
      async_copy_stage(gA, gB, cur ? lA0 : lA1, cur ? lB0 : lB1,
                       (s + 1) * KSTAGE);
      asm volatile("s_wait_asynccnt 8" ::: "memory");  // current 8 ops done
    } else {
      asm volatile("s_wait_asynccnt 0" ::: "memory");
    }
    __syncthreads();                                   // all waves' copies visible

    const unsigned short* Ash = &As[cur][0];
    const unsigned short* Bsh = &Bs[cur][0];
#pragma unroll
    for (int ks = 0; ks < 2; ++ks) {                   // two K=32 WMMA steps
      const int ko = ks * 32;
      Frag af[2], bfm[4];
#pragma unroll
      for (int mt = 0; mt < 2; ++mt) {
        const int row = wm + mt * 16 + fm;
        af[mt].h[0] = *(const v8us*)&Ash[row * LDSW + ko + ka];
        af[mt].h[1] = *(const v8us*)&Ash[row * LDSW + ko + ka + 16];
      }
#pragma unroll
      for (int nt = 0; nt < 4; ++nt) {
        const int col = wn + nt * 16 + fm;
        bfm[nt].h[0] = *(const v8us*)&Bsh[col * LDSW + ko + kb];
        bfm[nt].h[1] = *(const v8us*)&Bsh[col * LDSW + ko + kb + 16];
      }
#pragma unroll
      for (int mt = 0; mt < 2; ++mt)
#pragma unroll
        for (int nt = 0; nt < 4; ++nt)
          acc[mt][nt] = __builtin_amdgcn_wmma_f32_16x16x32_bf16(
              false, af[mt].v, false, bfm[nt].v, (short)0, acc[mt][nt],
              false, false);
    }
    __syncthreads();  // reads done before this buffer is refilled at s+1's issue
  }

  // epilogue: C layout = VGPR v, lanes 0-15 -> M=v, lanes 16-31 -> M=v+8; N=lane%16
  float biasr[4];
#pragma unroll
  for (int nt = 0; nt < 4; ++nt) biasr[nt] = bias[n0 + wn + nt * 16 + (lane & 15)];
  const int rbase = m0 + wm + ((lane >> 4) << 3);
#pragma unroll
  for (int mt = 0; mt < 2; ++mt)
#pragma unroll
    for (int nt = 0; nt < 4; ++nt) {
      const int col = n0 + wn + nt * 16 + (lane & 15);
#pragma unroll
      for (int v = 0; v < 8; ++v) {
        const int row = rbase + mt * 16 + v;
        out[(size_t)row * OUTD + col] = acc[mt][nt][v] + biasr[nt];
      }
    }
}

// ---------------------------------------------------------------------------
extern "C" void kernel_launch(void* const* d_in, const int* in_sizes, int n_in,
                              void* d_out, int out_size, void* d_ws, size_t ws_size,
                              hipStream_t stream) {
  const float*     data   = (const float*)d_in[0];
  const float*     w_real = (const float*)d_in[1];
  const float*     w_imag = (const float*)d_in[2];
  const float*     bias   = (const float*)d_in[3];
  const long long* zmat   = (const long long*)d_in[4];
  float*           out    = (float*)d_out;

  // workspace layout (all 16B aligned):
  //   T    : 4096 * 2049 float2  = 67,141,632 B
  //   Wbf  : 4096 * 4096 ushort  = 33,554,432 B
  //   Abf  : 2048 * 4096 ushort  = 16,777,216 B   (total ~118 MB)
  char* ws = (char*)d_ws;
  float2*         T   = (float2*)ws;
  unsigned short* Wbf = (unsigned short*)(ws + (size_t)OUTD * KH * sizeof(float2));
  unsigned short* Abf = (unsigned short*)(ws + (size_t)OUTD * KH * sizeof(float2)
                                             + (size_t)OUTD * IND * sizeof(unsigned short));

  // 1) activations -> bf16
  cast_bf16_kernel<<<2048, 256, 0, stream>>>(data, Abf, (BATCH * IND) / 4);

  // 2) masked inverse FFT along axis 0 (only first IN/2+1 columns needed)
  fft_col_kernel<<<KH, 256, 0, stream>>>(w_real, w_imag, zmat, T);

  // 3) irfft along axis 1 -> bf16 spatial weight
  fft_row_kernel<<<OUTD, 256, 0, stream>>>(T, Wbf);

  // 4) WMMA GEMM, async-LDS double-buffered, fused bias
  dim3 grid(OUTD / 128, BATCH / 128);
  gemm_wmma_bf16_kernel<<<grid, 256, 0, stream>>>(Abf, Wbf, bias, out);
}
